// NGen3OWLatentAttention_48670569398515
// MI455X (gfx1250) — matrix operand order repaired and checked by reading, not compiled, not run on hardware
//
#include <hip/hip_runtime.h>
#include <math.h>

// ---------------- problem constants ----------------
#define S_LEN   4096
#define HID     2048
#define NH      16
#define Q_LORA  768
#define KV_LORA 512
#define D_ROPE  64
#define D_V     128
#define TOP_K   1024
#define D_HEAD  192          // D_V + D_ROPE
#define EPSV    1e-6f

typedef unsigned short u16;
typedef __attribute__((ext_vector_type(16))) __bf16 v16bf;
typedef __attribute__((ext_vector_type(8)))  float  v8f;
typedef __attribute__((ext_vector_type(4)))  unsigned int v4u;
typedef __attribute__((ext_vector_type(8)))  unsigned int v8u;

// ---------------- scalar helpers ----------------
__device__ __forceinline__ u16 f2b(float x) {
    unsigned int u = __builtin_bit_cast(unsigned int, x);
    return (u16)((u + 0x7FFFu + ((u >> 16) & 1u)) >> 16);  // RNE
}
__device__ __forceinline__ float b2f(u16 h) {
    unsigned int u = ((unsigned int)h) << 16;
    return __builtin_bit_cast(float, u);
}

union FragCast { v8u u; v16bf b; };

// A-matrix fragment (16x32 bf16), row-major source, contiguous K.
// ISA layout: lanes 0-15 hold M=lane, K = {kk..kk+7, kk+16..kk+23};
//             lanes 16-31 hold M=lane-16, K = {kk+8..kk+15, kk+24..kk+31}.
__device__ __forceinline__ v16bf load_a_frag(const u16* base, int ld, int m0, int kk, int lane) {
    const u16* p = base + (size_t)(m0 + (lane & 15)) * ld + kk + ((lane >> 4) << 3);
    v4u lo = *(const v4u*)(p);        // K offsets +0..+7
    v4u hi = *(const v4u*)(p + 16);   // K offsets +16..+23
    FragCast f; f.u = __builtin_shufflevector(lo, hi, 0, 1, 2, 3, 4, 5, 6, 7);
    return f.b;
}

// B-matrix fragment (32x16 bf16) from Bt[N][K] row-major (i.e. column of B contiguous).
// ISA layout: lanes 0-15 hold N=lane, K = {kk..kk+7, kk+16..kk+23}; lanes 16-31 hold K = {kk+8..kk+15, kk+24..kk+31}.
__device__ __forceinline__ v16bf load_b_frag(const u16* baseT, int ld, int n0, int kk, int lane) {
    const u16* p = baseT + (size_t)(n0 + (lane & 15)) * ld + kk + ((lane >> 4) << 3);
    v4u lo = *(const v4u*)(p);
    v4u hi = *(const v4u*)(p + 16);
    FragCast f; f.u = __builtin_shufflevector(lo, hi, 0, 1, 2, 3, 4, 5, 6, 7);
    return f.b;
}

// A-fragment sourced from f32 (LDS softmax probs), converted on the fly.
__device__ __forceinline__ v16bf load_a_frag_f32(const float* base, int ld, int m0, int kk, int lane) {
    const float* p = base + (size_t)(m0 + (lane & 15)) * ld + kk + ((lane >> 4) << 3);
    u16 t[16];
#pragma unroll
    for (int j = 0; j < 8; ++j) { t[j] = f2b(p[j]); t[8 + j] = f2b(p[16 + j]); }
    v16bf out; __builtin_memcpy(&out, t, 32);
    return out;
}

__device__ __forceinline__ v8f wmma_bf16(v16bf a, v16bf b, v8f c) {
    return __builtin_amdgcn_wmma_f32_16x16x32_bf16(false, a, false, b, (short)0, c, false, false);
}

// ---------------- elementwise kernels ----------------
__global__ void k_cvt_bf16(const float* __restrict__ x, u16* __restrict__ y, int n) {
    int i = blockIdx.x * blockDim.x + threadIdx.x;
    if (i < n) y[i] = f2b(x[i]);
}

// W[K][N] f32 -> Wt[N][K] bf16
__global__ void k_transpose_bf16(const float* __restrict__ w, u16* __restrict__ wt, int K, int N) {
    int i = blockIdx.x * blockDim.x + threadIdx.x;
    if (i >= N * K) return;
    int n = i / K, k = i - n * K;
    wt[i] = f2b(w[(size_t)k * N + n]);
}

// gather hidden rows at dsa indices -> bf16 [TOP_K][HID]
__global__ void k_gather_rows(const float* __restrict__ h, const int* __restrict__ idx,
                              u16* __restrict__ out, int n) {
    int i = blockIdx.x * blockDim.x + threadIdx.x;
    if (i >= n) return;
    int j = i / HID, c = i - j * HID;
    out[i] = f2b(h[(size_t)idx[j] * HID + c]);
}

// RMSNorm: one block per row
__global__ void k_rmsnorm(const float* __restrict__ X, int ld, int cols,
                          const float* __restrict__ w, u16* __restrict__ Y, int oy) {
    __shared__ float red[256];
    int row = blockIdx.x, t = threadIdx.x;
    const float* x = X + (size_t)row * ld;
    float s = 0.f;
    for (int c = t; c < cols; c += 256) { float v = x[c]; s += v * v; }
    red[t] = s; __syncthreads();
    for (int off = 128; off; off >>= 1) { if (t < off) red[t] += red[t + off]; __syncthreads(); }
    float inv = rsqrtf(red[0] / (float)cols + EPSV);
    for (int c = t; c < cols; c += 256) Y[(size_t)row * oy + c] = f2b(w[c] * x[c] * inv);
}

__device__ __forceinline__ void yarn_cos_sin(int pos, int i, float* c, float* s) {
    const float ms = 0.1f * logf(32.0f) + 1.0f;
    const float ms2 = ms * ms;
    float inv_freq = expf(-((float)i / 32.0f) * logf(1.0e7f)) / 32.0f;
    float f = ((float)pos / 32.0f) * inv_freq;
    *c = cosf(f) * ms2; *s = sinf(f) * ms2;
}

// q_full bf16 [NH][S][192] from q bf16 [S][NH*128] and q_rope bf16 [S][NH*64]
__global__ void k_rope_q(const u16* __restrict__ q, const u16* __restrict__ qr,
                         const int* __restrict__ pos_ids, u16* __restrict__ qf, int n) {
    int i = blockIdx.x * blockDim.x + threadIdx.x;
    if (i >= n) return;
    int h = i / (S_LEN * D_HEAD);
    int rem = i - h * (S_LEN * D_HEAD);
    int s = rem / D_HEAD, d = rem - s * D_HEAD;
    if (d < D_V) { qf[i] = q[(size_t)s * (NH * D_V) + h * D_V + d]; return; }
    int dr = d - D_V;
    const u16* qrow = qr + (size_t)s * (NH * D_ROPE) + h * D_ROPE;
    float x  = b2f(qrow[dr]);
    float x2 = (dr < 32) ? -b2f(qrow[dr + 32]) : b2f(qrow[dr - 32]);
    float c, sn; yarn_cos_sin(pos_ids[s], dr & 31, &c, &sn);
    qf[i] = f2b(x * c + x2 * sn);
}

// k_full bf16 [NH][TOP_K][192] from kv bf16 [TOP_K][NH*256] and kv_lr f32 rope part
__global__ void k_build_k(const u16* __restrict__ kv, const float* __restrict__ kvlr,
                          const int* __restrict__ pos_ids, const int* __restrict__ idx,
                          u16* __restrict__ kf, int n) {
    int i = blockIdx.x * blockDim.x + threadIdx.x;
    if (i >= n) return;
    int h = i / (TOP_K * D_HEAD);
    int rem = i - h * (TOP_K * D_HEAD);
    int j = rem / D_HEAD, d = rem - j * D_HEAD;
    if (d < D_V) { kf[i] = kv[(size_t)j * (NH * 2 * D_V) + h * (2 * D_V) + d]; return; }
    int dr = d - D_V;
    const float* krow = kvlr + (size_t)j * (KV_LORA + D_ROPE) + KV_LORA;
    float x  = krow[dr];
    float x2 = (dr < 32) ? -krow[dr + 32] : krow[dr - 32];
    float c, sn; yarn_cos_sin(pos_ids[idx[j]], dr & 31, &c, &sn);
    kf[i] = f2b(x * c + x2 * sn);
}

// V transposed: vt bf16 [NH][128][TOP_K]
__global__ void k_build_vt(const u16* __restrict__ kv, u16* __restrict__ vt, int n) {
    int i = blockIdx.x * blockDim.x + threadIdx.x;
    if (i >= n) return;
    int h = i / (D_V * TOP_K);
    int rem = i - h * (D_V * TOP_K);
    int d = rem / TOP_K, j = rem - d * TOP_K;
    vt[i] = kv[(size_t)j * (NH * 2 * D_V) + h * (2 * D_V) + D_V + d];
}

// ---------------- NT GEMM: C[M][N] = A[M][K](bf16) * Bt[N][K]^T(bf16) ----------------
// one wave per block, 64(M) x 32(N) register tile
__global__ __launch_bounds__(32) void k_gemm_nt(const u16* __restrict__ A, const u16* __restrict__ Bt,
                                                void* __restrict__ C, int M, int N, int K, int out_bf16) {
    int lane = threadIdx.x;
    int n0 = blockIdx.x * 32;
    int m0 = blockIdx.y * 64;
    v8f acc[4][2] = {};
    for (int kk = 0; kk < K; kk += 32) {
        v16bf a[4], b[2];
#pragma unroll
        for (int i = 0; i < 4; ++i) a[i] = load_a_frag(A, K, m0 + 16 * i, kk, lane);
#pragma unroll
        for (int j = 0; j < 2; ++j) b[j] = load_b_frag(Bt, K, n0 + 16 * j, kk, lane);
        if (kk + 32 < K) {
            __builtin_prefetch(A + (size_t)(m0 + (lane & 15)) * K + kk + 32, 0, 3);
            __builtin_prefetch(Bt + (size_t)(n0 + (lane & 15)) * K + kk + 32, 0, 3);
        }
#pragma unroll
        for (int i = 0; i < 4; ++i)
#pragma unroll
            for (int j = 0; j < 2; ++j) acc[i][j] = wmma_bf16(a[i], b[j], acc[i][j]);
    }
    int ln = lane & 15, mh = (lane >> 4) * 8;
#pragma unroll
    for (int i = 0; i < 4; ++i)
#pragma unroll
        for (int j = 0; j < 2; ++j) {
            int col = n0 + 16 * j + ln;
#pragma unroll
            for (int r = 0; r < 8; ++r) {
                size_t off = (size_t)(m0 + 16 * i + mh + r) * N + col;
                if (out_bf16) ((u16*)C)[off] = f2b(acc[i][j][r]);
                else          ((float*)C)[off] = acc[i][j][r];
            }
        }
}

// ---------------- attention: 4 waves per (head, 16-query tile) ----------------
// Waves split the 64 key tiles for QK^T, cooperate on softmax (8 lanes/row),
// then split the 8 D_V chunks for P.V. One shared 64KB score buffer.
__global__ __launch_bounds__(128) void k_attention(const u16* __restrict__ qf, const u16* __restrict__ kf,
                                                   const u16* __restrict__ vt, u16* __restrict__ attn,
                                                   float scale) {
    __shared__ float sc[16 * TOP_K];   // 64 KB score/prob buffer
    __shared__ float red[16][8];
    int tid  = threadIdx.x;
    int lane = tid & 31;
    int wave = tid >> 5;               // 0..3
    int h  = blockIdx.x >> 8;          // / (S/16)
    int qt = blockIdx.x & 255;
    int m0 = qt * 16;
    const u16* A = qf + (size_t)h * S_LEN * D_HEAD;   // [S][192]
    const u16* B = kf + (size_t)h * TOP_K * D_HEAD;   // [TOP_K][192]
    int ln = lane & 15, mh = (lane >> 4) * 8;

    // preload 6 A-fragments (K = 192); same for all 4 waves (L0/L2 hit)
    v16bf afr[6];
#pragma unroll
    for (int kk6 = 0; kk6 < 6; ++kk6) afr[kk6] = load_a_frag(A, D_HEAD, m0, kk6 * 32, lane);

    // scores = q . k^T, scaled, into LDS; key tiles strided across waves
    for (int nt = wave; nt < TOP_K / 16; nt += 4) {
        v8f c = {};
#pragma unroll
        for (int kk6 = 0; kk6 < 6; ++kk6)
            c = wmma_bf16(afr[kk6], load_b_frag(B, D_HEAD, nt * 16, kk6 * 32, lane), c);
#pragma unroll
        for (int r = 0; r < 8; ++r) sc[(mh + r) * TOP_K + nt * 16 + ln] = c[r] * scale;
    }
    __syncthreads();

    // softmax: 8 threads per row, 128-element chunks, LDS tree reduction
    {
        int r  = tid >> 3;             // row 0..15
        int sb = tid & 7;              // sub-chunk 0..7
        float* row = sc + r * TOP_K;
        int k0 = sb * 128, k1 = k0 + 128;
        float mx = -3.4e38f;
        for (int k = k0; k < k1; ++k) mx = fmaxf(mx, row[k]);
        red[r][sb] = mx; __syncthreads();
        if (sb == 0) {
            float m = red[r][0];
#pragma unroll
            for (int j = 1; j < 8; ++j) m = fmaxf(m, red[r][j]);
            red[r][0] = m;
        }
        __syncthreads();
        mx = red[r][0];
        __syncthreads();
        float s = 0.f;
        for (int k = k0; k < k1; ++k) { float e = __expf(row[k] - mx); row[k] = e; s += e; }
        red[r][sb] = s; __syncthreads();
        if (sb == 0) {
            float t = 0.f;
#pragma unroll
            for (int j = 0; j < 8; ++j) t += red[r][j];
            red[r][0] = t;
        }
        __syncthreads();
        float inv = 1.0f / red[r][0];
        for (int k = k0; k < k1; ++k) row[k] *= inv;
    }
    __syncthreads();

    // out = P . V  (P: 16 x 1024 in LDS f32, Vt: [128][1024] bf16); chunks split by wave
    const u16* Vt = vt + (size_t)h * D_V * TOP_K;
#pragma unroll
    for (int q = 0; q < 2; ++q) {
        int dt = wave * 2 + q;         // 0..7
        v8f c = {};
        for (int kk = 0; kk < TOP_K; kk += 32)
            c = wmma_bf16(load_a_frag_f32(sc, TOP_K, 0, kk, lane),
                          load_b_frag(Vt, TOP_K, dt * 16, kk, lane), c);
#pragma unroll
        for (int r = 0; r < 8; ++r)
            attn[(size_t)(m0 + mh + r) * (NH * D_V) + h * D_V + dt * 16 + ln] = f2b(c[r]);
    }
}

// ---------------- host launcher ----------------
static inline void* bump(char*& p, size_t bytes) {
    void* r = p; p += (bytes + 255) & ~(size_t)255; return r;
}

extern "C" void kernel_launch(void* const* d_in, const int* in_sizes, int n_in,
                              void* d_out, int out_size, void* d_ws, size_t ws_size,
                              hipStream_t stream) {
    (void)in_sizes; (void)n_in; (void)out_size; (void)ws_size;
    const float* hidden  = (const float*)d_in[0];
    const int*   pos_ids = (const int*)d_in[1];
    const int*   idx     = (const int*)d_in[2];
    const float* wq_a    = (const float*)d_in[3];
    const float* q_ln    = (const float*)d_in[4];
    const float* wq_b    = (const float*)d_in[5];
    const float* wq_rope = (const float*)d_in[6];
    const float* wkv_a   = (const float*)d_in[7];
    const float* kv_ln   = (const float*)d_in[8];
    const float* wkv_b   = (const float*)d_in[9];
    const float* wo      = (const float*)d_in[10];

    char* wp = (char*)d_ws;
    u16* hb     = (u16*)bump(wp, (size_t)S_LEN * HID * 2);
    u16* wqa_t  = (u16*)bump(wp, (size_t)Q_LORA * HID * 2);
    u16* wqb_t  = (u16*)bump(wp, (size_t)(NH * D_V) * Q_LORA * 2);
    u16* wqr_t  = (u16*)bump(wp, (size_t)(NH * D_ROPE) * Q_LORA * 2);
    u16* wkva_t = (u16*)bump(wp, (size_t)(KV_LORA + D_ROPE) * HID * 2);
    u16* wkvb_t = (u16*)bump(wp, (size_t)(NH * 2 * D_V) * KV_LORA * 2);
    u16* wo_t   = (u16*)bump(wp, (size_t)HID * (NH * D_V) * 2);
    u16* hg     = (u16*)bump(wp, (size_t)TOP_K * HID * 2);
    float* ql_raw = (float*)bump(wp, (size_t)S_LEN * Q_LORA * 4);
    u16* ql_b   = (u16*)bump(wp, (size_t)S_LEN * Q_LORA * 2);
    u16* q_b    = (u16*)bump(wp, (size_t)S_LEN * NH * D_V * 2);
    u16* qr_b   = (u16*)bump(wp, (size_t)S_LEN * NH * D_ROPE * 2);
    float* kvlr = (float*)bump(wp, (size_t)TOP_K * (KV_LORA + D_ROPE) * 4);
    u16* kvl_b  = (u16*)bump(wp, (size_t)TOP_K * KV_LORA * 2);
    u16* kv_b   = (u16*)bump(wp, (size_t)TOP_K * NH * 2 * D_V * 2);
    u16* qf     = (u16*)bump(wp, (size_t)NH * S_LEN * D_HEAD * 2);
    u16* kf     = (u16*)bump(wp, (size_t)NH * TOP_K * D_HEAD * 2);
    u16* vt     = (u16*)bump(wp, (size_t)NH * D_V * TOP_K * 2);
    u16* attn_b = (u16*)bump(wp, (size_t)S_LEN * NH * D_V * 2);

    const int T = 256;
    auto g = [](int n, int t) { return (n + t - 1) / t; };

    // conversions / transposes / gather
    k_cvt_bf16<<<g(S_LEN * HID, T), T, 0, stream>>>(hidden, hb, S_LEN * HID);
    k_transpose_bf16<<<g(HID * Q_LORA, T), T, 0, stream>>>(wq_a, wqa_t, HID, Q_LORA);
    k_transpose_bf16<<<g(Q_LORA * NH * D_V, T), T, 0, stream>>>(wq_b, wqb_t, Q_LORA, NH * D_V);
    k_transpose_bf16<<<g(Q_LORA * NH * D_ROPE, T), T, 0, stream>>>(wq_rope, wqr_t, Q_LORA, NH * D_ROPE);
    k_transpose_bf16<<<g(HID * (KV_LORA + D_ROPE), T), T, 0, stream>>>(wkv_a, wkva_t, HID, KV_LORA + D_ROPE);
    k_transpose_bf16<<<g(KV_LORA * NH * 2 * D_V, T), T, 0, stream>>>(wkv_b, wkvb_t, KV_LORA, NH * 2 * D_V);
    k_transpose_bf16<<<g(HID * HID, T), T, 0, stream>>>(wo, wo_t, NH * D_V, HID);
    k_gather_rows<<<g(TOP_K * HID, T), T, 0, stream>>>(hidden, idx, hg, TOP_K * HID);

    // q path
    k_gemm_nt<<<dim3(Q_LORA / 32, S_LEN / 64), 32, 0, stream>>>(hb, wqa_t, ql_raw, S_LEN, Q_LORA, HID, 0);
    k_rmsnorm<<<S_LEN, 256, 0, stream>>>(ql_raw, Q_LORA, Q_LORA, q_ln, ql_b, Q_LORA);
    k_gemm_nt<<<dim3((NH * D_V) / 32, S_LEN / 64), 32, 0, stream>>>(ql_b, wqb_t, q_b, S_LEN, NH * D_V, Q_LORA, 1);
    k_gemm_nt<<<dim3((NH * D_ROPE) / 32, S_LEN / 64), 32, 0, stream>>>(ql_b, wqr_t, qr_b, S_LEN, NH * D_ROPE, Q_LORA, 1);

    // kv path (only the TOP_K gathered rows)
    k_gemm_nt<<<dim3((KV_LORA + D_ROPE) / 32, TOP_K / 64), 32, 0, stream>>>(hg, wkva_t, kvlr, TOP_K, KV_LORA + D_ROPE, HID, 0);
    k_rmsnorm<<<TOP_K, 256, 0, stream>>>(kvlr, KV_LORA + D_ROPE, KV_LORA, kv_ln, kvl_b, KV_LORA);
    k_gemm_nt<<<dim3((NH * 2 * D_V) / 32, TOP_K / 64), 32, 0, stream>>>(kvl_b, wkvb_t, kv_b, TOP_K, NH * 2 * D_V, KV_LORA, 1);

    // RoPE + packing
    k_rope_q<<<g(NH * S_LEN * D_HEAD, T), T, 0, stream>>>(q_b, qr_b, pos_ids, qf, NH * S_LEN * D_HEAD);
    k_build_k<<<g(NH * TOP_K * D_HEAD, T), T, 0, stream>>>(kv_b, kvlr, pos_ids, idx, kf, NH * TOP_K * D_HEAD);
    k_build_vt<<<g(NH * D_V * TOP_K, T), T, 0, stream>>>(kv_b, vt, NH * D_V * TOP_K);

    // attention
    float scale = 1.0f / sqrtf((float)D_HEAD);
    k_attention<<<NH * (S_LEN / 16), 128, 0, stream>>>(qf, kf, vt, attn_b, scale);

    // output projection -> d_out (f32)
    k_gemm_nt<<<dim3(HID / 32, S_LEN / 64), 32, 0, stream>>>(attn_b, wo_t, (float*)d_out, S_LEN, HID, NH * D_V, 0);
}
